// EquivariantAttentionLayer_67946382623007
// MI455X (gfx1250) — compile-verified, matrix operand order checked by reading, not run on hardware
//
#include <hip/hip_runtime.h>

typedef __attribute__((ext_vector_type(16))) __bf16 v16bf;
typedef __attribute__((ext_vector_type(8)))  __bf16 v8bf;
typedef __attribute__((ext_vector_type(8)))  float  v8f;
typedef __attribute__((ext_vector_type(4)))  int    v4i;

// ---------------------------------------------------------------------------
// CDNA5 async global->LDS copy (ASYNCcnt path). Guarded so the file compiles
// on toolchains without the builtin (falls back to b128 VGPR copies).
// Probe result: param 0 is v4i in address space 1, param 1 in address space 3.
// ---------------------------------------------------------------------------
#if __has_builtin(__builtin_amdgcn_global_load_async_to_lds_b128) && \
    __has_builtin(__builtin_amdgcn_s_wait_asynccnt)
#define HAVE_ASYNC_LDS 1
#endif

static __device__ inline void cp16(void* lds, const void* g) {
#ifdef HAVE_ASYNC_LDS
    auto gp = (__attribute__((address_space(1))) v4i*)(unsigned long long)g;
    auto lp = (__attribute__((address_space(3))) v4i*)(unsigned int)(unsigned long long)lds;
    __builtin_amdgcn_global_load_async_to_lds_b128(gp, lp, 0, 0);
#else
    *(v8bf*)lds = *(const v8bf*)g;
#endif
}
static __device__ inline void cp_fence() {
#ifdef HAVE_ASYNC_LDS
    __builtin_amdgcn_s_wait_asynccnt(0);
#endif
}

// ---------------------------------------------------------------------------
// Fragment load from LDS for 16-bit WMMA operands (A: 16xK rows, B stored as
// [n][k] row-major).  ISA 16-bit layout: lanes 0-15 hold K=0..7,16..23,
// lanes 16-31 hold K=8..15,24..31 (two contiguous 16-byte chunks per lane).
// ---------------------------------------------------------------------------
static __device__ inline v16bf frag_ld(const __bf16* row0, int ldr, int lane) {
    const int r = lane & 15, half = (lane >> 4) & 1;
    const __bf16* p = row0 + r * ldr + half * 8;
    v8bf c0 = *(const v8bf*)(p);
    v8bf c1 = *(const v8bf*)(p + 16);
    return __builtin_shufflevector(c0, c1, 0,1,2,3,4,5,6,7,8,9,10,11,12,13,14,15);
}

static __device__ inline float ldf(const float* p)  { return *p; }
static __device__ inline float ldf(const __bf16* p) { return (float)*p; }

#define WMMA_BF16(a, b, c) \
    __builtin_amdgcn_wmma_f32_16x16x32_bf16(false, (a), false, (b), (short)0, (c), false, false)

// ---------------------------------------------------------------------------
// Generic tiled GEMM:  C[m][n] = sum_k A[m][k] * B[k][n]   (bf16 WMMA, f32 acc)
// A element at A[m*rsA + k*csA], B at B[k*rsB + n*csB]; blockIdx.z selects the
// w-matrix via wsA/wsB/wsC offsets.
// Block tile 128x128, 8 waves of 32x64, BK=32, double-buffered LDS with
// register staging (loads for tile k+1 overlap the WMMAs of tile k).
// ---------------------------------------------------------------------------
template <typename TA, typename TB>
__global__ __launch_bounds__(256)
void gemm_wmma_bf16(const TA* __restrict__ A, long rsA, long csA, long wsA,
                    const TB* __restrict__ B, long rsB, long csB, long wsB,
                    __bf16* __restrict__ C, long rsC, long wsC,
                    int K)
{
    __shared__ __bf16 As[2][128][40];   // [buf][m][k], 80B row stride (16B aligned)
    __shared__ __bf16 Bs[2][128][40];   // [buf][n][k] (k-major per n)

    A += (long)blockIdx.z * wsA;
    B += (long)blockIdx.z * wsB;
    C += (long)blockIdx.z * wsC;

    const int tid  = threadIdx.x;
    const int lane = tid & 31;
    const int wv   = tid >> 5;            // 0..7
    const int m0   = blockIdx.x * 128;
    const int n0   = blockIdx.y * 128;
    const int wm   = (wv >> 1) * 32;      // 0,32,64,96
    const int wn   = (wv & 1) * 64;       // 0,64

    // per-thread element coordinates for the 128x32 A tile / 32x128 B tile
    int rA[16], kA[16], rB[16], kB[16];
    #pragma unroll
    for (int t = 0; t < 16; ++t) {
        int i = tid + t * 256;
        if (csA == 1) { rA[t] = i >> 5;  kA[t] = i & 31; }   // row-major: k fastest
        else          { rA[t] = i & 127; kA[t] = i >> 7; }   // col-major: r fastest
        if (csB == 1) { rB[t] = i & 127; kB[t] = i >> 7; }   // row-major: n fastest
        else          { rB[t] = i >> 5;  kB[t] = i & 31; }   // col-major: k fastest
    }

    float ar[16], br[16];
    #pragma unroll
    for (int t = 0; t < 16; ++t) {       // prologue: fetch K-tile 0
        ar[t] = ldf(&A[(long)(m0 + rA[t]) * rsA + (long)kA[t] * csA]);
        br[t] = ldf(&B[(long)kB[t] * rsB + (long)(n0 + rB[t]) * csB]);
    }
    #pragma unroll
    for (int t = 0; t < 16; ++t) {
        As[0][rA[t]][kA[t]] = (__bf16)ar[t];
        Bs[0][rB[t]][kB[t]] = (__bf16)br[t];
    }

    v8f acc[2][4] = {};
    const int KT = K >> 5;

    for (int kt = 0; kt < KT; ++kt) {
        __syncthreads();
        const int cur = kt & 1;

        if (kt + 1 < KT) {               // issue global loads for next tile
            const int k0 = (kt + 1) << 5;
            #pragma unroll
            for (int t = 0; t < 16; ++t) {
                ar[t] = ldf(&A[(long)(m0 + rA[t]) * rsA + (long)(k0 + kA[t]) * csA]);
                br[t] = ldf(&B[(long)(k0 + kB[t]) * rsB + (long)(n0 + rB[t]) * csB]);
            }
        }

        v16bf a0 = frag_ld(&As[cur][wm][0],      40, lane);
        v16bf a1 = frag_ld(&As[cur][wm + 16][0], 40, lane);
        #pragma unroll
        for (int tj = 0; tj < 4; ++tj) {
            v16bf b = frag_ld(&Bs[cur][wn + tj * 16][0], 40, lane);
            acc[0][tj] = WMMA_BF16(a0, b, acc[0][tj]);
            acc[1][tj] = WMMA_BF16(a1, b, acc[1][tj]);
        }

        if (kt + 1 < KT) {               // LDS stores land after the WMMAs
            #pragma unroll
            for (int t = 0; t < 16; ++t) {
                As[cur ^ 1][rA[t]][kA[t]] = (__bf16)ar[t];
                Bs[cur ^ 1][rB[t]][kB[t]] = (__bf16)br[t];
            }
        }
    }

    const int half = lane >> 4, cl = lane & 15;
    #pragma unroll
    for (int ti = 0; ti < 2; ++ti)
      #pragma unroll
      for (int tj = 0; tj < 4; ++tj)
        #pragma unroll
        for (int v = 0; v < 8; ++v) {
            int row = m0 + wm + ti * 16 + half * 8 + v;
            int col = n0 + wn + tj * 16 + cl;
            C[(long)row * rsC + col] = (__bf16)acc[ti][tj][v];
        }
}

// ---------------------------------------------------------------------------
// Temporal attention: one block per (j point, h head).
//   S[a][b] = q[a]·k[b] (128x128, K=32) ; row-softmax over b ;
//   out[i][m] = sum_I  S_sm[I][i] * V[I][m]   (transposed-weights contraction,
//   faithful to the reference's 'hIij' label swap).
// qkv layout: (w, i, j, h, m) bf16.  ta layout: (h, m, i, j) bf16.
// ---------------------------------------------------------------------------
__global__ __launch_bounds__(256)
void temporal_attn(const __bf16* __restrict__ qkv, __bf16* __restrict__ ta)
{
    extern __shared__ char smem[];
    float*  S  = (float*)smem;                        // [128][132] f32  (67584 B)
    __bf16* Qs = (__bf16*)(smem + 67584);             // [128][40]
    __bf16* Ks = (__bf16*)(smem + 67584 + 10240);     // [128][40]  (= B[n=I][k=m])
    __bf16* Vt = (__bf16*)(smem + 67584 + 20480);     // [32][136]  (= B[n=m][k=I])

    const int j = blockIdx.x;          // 0..255
    const int h = blockIdx.y;          // 0..15
    const int tid = threadIdx.x, lane = tid & 31, wv = tid >> 5;

    const long strideI = 256L * 512;   // +1 frame
    const long strideW = 128L * strideI;
    const long base = (long)j * 512 + h * 32;

    // Q, K: straight rows -> async 16B copies (512 chunks each)
    #pragma unroll
    for (int t = 0; t < 2; ++t) {
        int c = tid + t * 256;                 // 0..511
        int i = c >> 2, m8 = (c & 3) * 8;
        long g = base + (long)i * strideI + m8;
        cp16(&Qs[i * 40 + m8], &qkv[g]);
        cp16(&Ks[i * 40 + m8], &qkv[g + strideW]);
    }
    // V: transposed into LDS -> VGPR path
    #pragma unroll
    for (int t = 0; t < 16; ++t) {
        int idx = tid + t * 256;
        int i = idx >> 5, m = idx & 31;
        Vt[m * 136 + i] = qkv[base + (long)i * strideI + m + 2 * strideW];
    }
    cp_fence();
    __syncthreads();

    // ---- S = Q K^T : wave wv -> rows [wv*16, wv*16+16), 8 column tiles -----
    {
        v16bf a = frag_ld(&Qs[(wv * 16) * 40], 40, lane);
        const int half = lane >> 4, cl = lane & 15;
        #pragma unroll
        for (int c = 0; c < 8; ++c) {
            v16bf b = frag_ld(&Ks[(c * 16) * 40], 40, lane);
            v8f acc = {};
            acc = WMMA_BF16(a, b, acc);
            #pragma unroll
            for (int v = 0; v < 8; ++v)
                S[(wv * 16 + half * 8 + v) * 132 + c * 16 + cl] = acc[v];
        }
    }
    __syncthreads();

    // ---- row softmax over 128 columns --------------------------------------
    if (tid < 128) {
        float* row = S + tid * 132;
        float mx = row[0];
        for (int c = 1; c < 128; ++c) mx = fmaxf(mx, row[c]);
        float sum = 0.f;
        for (int c = 0; c < 128; ++c) { float e = __expf(row[c] - mx); row[c] = e; sum += e; }
        float inv = 1.f / sum;
        for (int c = 0; c < 128; ++c) row[c] *= inv;
    }
    __syncthreads();

    // ---- out = W^T V : A[i][I] = S_sm[I][i] (transposed gather) ------------
    v8f acc[2] = {};
    const int r = lane & 15, half = lane >> 4;
    #pragma unroll
    for (int kk = 0; kk < 128; kk += 32) {
        v16bf a;
        #pragma unroll
        for (int e = 0; e < 16; ++e) {
            int I = kk + (e & 7) + (e >> 3) * 16 + half * 8;
            a[e] = (__bf16)S[I * 132 + wv * 16 + r];
        }
        v16bf b0 = frag_ld(&Vt[0 * 136 + kk],  136, lane);
        v16bf b1 = frag_ld(&Vt[16 * 136 + kk], 136, lane);
        acc[0] = WMMA_BF16(a, b0, acc[0]);
        acc[1] = WMMA_BF16(a, b1, acc[1]);
    }
    #pragma unroll
    for (int tj = 0; tj < 2; ++tj)
      #pragma unroll
      for (int v = 0; v < 8; ++v) {
          int i = wv * 16 + half * 8 + v;
          int m = tj * 16 + (lane & 15);
          ta[((long)(h * 32 + m) * 128 + i) * 256 + j] = (__bf16)acc[tj][v];
      }
}

// ---------------------------------------------------------------------------
// Point attention: one block per (j-half, frame i, head h).
//   S[jq][J] = q[jq]·k[J] (128x256, K=32); softmax over J; out = W·V (standard).
// qkv layout: (w, i, j, h, m) bf16.  out layout: (i, j, h*M+m) f32.
// ---------------------------------------------------------------------------
__global__ __launch_bounds__(256)
void point_attn(const __bf16* __restrict__ qkv, float* __restrict__ out)
{
    extern __shared__ char smem[];
    float*  S  = (float*)smem;                 // [128][260] f32   (133120 B)
    __bf16* Qs = (__bf16*)(smem + 133120);     // [128][40]
    __bf16* Ks = (__bf16*)(smem + 143360);     // [256][40]  (= B[n=J][k=m])
    __bf16* Vt = (__bf16*)(smem + 163840);     // [32][264]  (= B[n=m][k=J])

    const int jb = blockIdx.x;   // 0..1
    const int i  = blockIdx.y;   // 0..127
    const int h  = blockIdx.z;   // 0..15
    const int tid = threadIdx.x, lane = tid & 31, wv = tid >> 5;

    const long strideW = 128L * 256 * 512;
    const long base = ((long)i * 256) * 512 + h * 32;

    // Q (128 rows of this half) and K (all 256 rows): async 16B copies
    #pragma unroll
    for (int t = 0; t < 2; ++t) {
        int c = tid + t * 256;                 // 0..511
        int r = c >> 2, m8 = (c & 3) * 8;
        cp16(&Qs[r * 40 + m8], &qkv[base + (long)(jb * 128 + r) * 512 + m8]);
    }
    #pragma unroll
    for (int t = 0; t < 4; ++t) {
        int c = tid + t * 256;                 // 0..1023
        int r = c >> 2, m8 = (c & 3) * 8;
        cp16(&Ks[r * 40 + m8], &qkv[base + (long)r * 512 + m8 + strideW]);
    }
    // V transposed -> VGPR path
    #pragma unroll
    for (int t = 0; t < 32; ++t) {
        int idx = tid + t * 256; int r = idx >> 5, m = idx & 31;
        Vt[m * 264 + r] = qkv[base + (long)r * 512 + m + 2 * strideW];
    }
    cp_fence();
    __syncthreads();

    // ---- S = Q K^T : wave wv -> 16 rows x 256 cols -------------------------
    {
        v16bf a = frag_ld(&Qs[(wv * 16) * 40], 40, lane);
        const int half = lane >> 4, cl = lane & 15;
        #pragma unroll
        for (int c = 0; c < 16; ++c) {
            v16bf b = frag_ld(&Ks[(c * 16) * 40], 40, lane);
            v8f acc = {};
            acc = WMMA_BF16(a, b, acc);
            #pragma unroll
            for (int v = 0; v < 8; ++v)
                S[(wv * 16 + half * 8 + v) * 260 + c * 16 + cl] = acc[v];
        }
    }
    __syncthreads();

    if (tid < 128) {                           // row softmax over 256 cols
        float* row = S + tid * 260;
        float mx = row[0];
        for (int c = 1; c < 256; ++c) mx = fmaxf(mx, row[c]);
        float sum = 0.f;
        for (int c = 0; c < 256; ++c) { float e = __expf(row[c] - mx); row[c] = e; sum += e; }
        float inv = 1.f / sum;
        for (int c = 0; c < 256; ++c) row[c] *= inv;
    }
    __syncthreads();

    // ---- out = W V : straight-read A, K loop over J ------------------------
    v8f acc[2] = {};
    const int rl = lane & 15, half = lane >> 4;
    #pragma unroll
    for (int kk = 0; kk < 256; kk += 32) {
        v16bf a;
        #pragma unroll
        for (int e = 0; e < 16; ++e) {
            int J = kk + (e & 7) + (e >> 3) * 16 + half * 8;
            a[e] = (__bf16)S[(wv * 16 + rl) * 260 + J];
        }
        v16bf b0 = frag_ld(&Vt[0 * 264 + kk],  264, lane);
        v16bf b1 = frag_ld(&Vt[16 * 264 + kk], 264, lane);
        acc[0] = WMMA_BF16(a, b0, acc[0]);
        acc[1] = WMMA_BF16(a, b1, acc[1]);
    }
    #pragma unroll
    for (int tj = 0; tj < 2; ++tj)
      #pragma unroll
      for (int v = 0; v < 8; ++v) {
          int jq = jb * 128 + wv * 16 + half * 8 + v;
          int m  = tj * 16 + (lane & 15);
          out[((long)i * 256 + jq) * 512 + h * 32 + m] = acc[tj][v];
      }
}

// ---------------------------------------------------------------------------
extern "C" void kernel_launch(void* const* d_in, const int* in_sizes, int n_in,
                              void* d_out, int out_size, void* d_ws, size_t ws_size,
                              hipStream_t stream)
{
    const float* x   = (const float*)d_in[0];   // (128, 256, 256)
    const float* wqt = (const float*)d_in[1];   // (3, 256, 16, 32)
    const float* wqp = (const float*)d_in[2];   // (3, 16, 32, 16, 32)
    float* out = (float*)d_out;                 // (128, 256, 512) f32

    __bf16* qkvb = (__bf16*)d_ws;               // 3 * 32768 * 512 bf16 (100.7 MB)
    __bf16* ta   = qkvb + 3L * 32768 * 512;     // 512 * 32768 bf16    (33.5 MB)

    // Stage 1: qkv[w][ij][hm] = x[ij][d] * Wt[w][d][hm]   (M=32768,N=512,K=256)
    {
        dim3 grid(32768 / 128, 512 / 128, 3), block(256);
        gemm_wmma_bf16<float, float><<<grid, block, 0, stream>>>(
            x,    256, 1, 0,
            wqt,  512, 1, 256L * 512,
            qkvb, 512, 32768L * 512,
            256);
    }

    // Stage 2: temporal attention -> ta (h, m, i, j) bf16
    temporal_attn<<<dim3(256, 16), 256, 96768, stream>>>(qkvb, ta);

    // Stage 3: qkv2[w][ij][hm] = ta[gn][ij] * Wp[w][hm][gn]
    //          A = ta^T (col-major), B = Wp^T (col-major); M=32768,N=512,K=512
    {
        dim3 grid(32768 / 128, 512 / 128, 3), block(256);
        gemm_wmma_bf16<__bf16, float><<<grid, block, 0, stream>>>(
            ta,   1, 32768, 0,
            wqp,  1, 512,   512L * 512,
            qkvb, 512, 32768L * 512,
            512);
    }

    // Stage 4: point attention -> out (i, j, h*32+m) f32
    point_attn<<<dim3(2, 128, 16), 256, 180736, stream>>>(qkvb, out);
}
